// MultiHeadAttention_58102317580888
// MI455X (gfx1250) — compile-verified, matrix operand order checked by reading, not run on hardware
//
#include <hip/hip_runtime.h>
#include <hip/hip_bf16.h>

// ---------------- CDNA5 WMMA types ----------------
typedef __bf16 bf16;
typedef __bf16 v16bf __attribute__((ext_vector_type(16)));
typedef float  v8f   __attribute__((ext_vector_type(8)));

static __device__ __forceinline__ v8f wmma_bf16(v16bf a, v16bf b, v8f c) {
  // D = A(16x32 bf16) x B(32x16 bf16) + C(16x16 f32)
  return __builtin_amdgcn_wmma_f32_16x16x32_bf16(
      /*neg_a=*/false, a, /*neg_b=*/false, b,
      /*c_mod=*/(short)0, c, /*reuse_a=*/false, /*reuse_b=*/false);
}

// ---------------- CDNA5 async copy to LDS ----------------
typedef __attribute__((address_space(3))) char lds_char_t;

static __device__ __forceinline__ unsigned lds_addr_of(const void* p) {
  return (unsigned)(unsigned long long)(lds_char_t*)p;
}

// GLOBAL_LOAD_ASYNC_TO_LDS_B128: 16 bytes per lane, tracked by ASYNCcnt
static __device__ __forceinline__ void async_ld16(unsigned lds_off, const void* g) {
  asm volatile("global_load_async_to_lds_b128 %0, %1, off"
               :: "v"(lds_off), "v"((unsigned long long)g)
               : "memory");
}

// async loads complete in order: waiting <=2 guarantees the older tile's
// two copies have landed while the next tile's two are still in flight.
static __device__ __forceinline__ void wait_asynccnt2() {
  asm volatile("s_wait_asynccnt 2" ::: "memory");
}
static __device__ __forceinline__ void wait_asynccnt0() {
  asm volatile("s_wait_asynccnt 0" ::: "memory");
}

#define BB 4
#define SS 2048
#define DD 512
#define HH 8
#define DH 64
#define MTOT (BB * SS)   // 8192

// ------------------------------------------------------------------
// Kernel 0: fold W_gen_S into W_K (per head) and convert all weights
// to bf16. W_Kg[d, h*64+j] = sum_i W_K[d, h*64+i] * W_gen_S[i, j]
// ------------------------------------------------------------------
__global__ __launch_bounds__(256) void prep_weights(
    const float* __restrict__ WQ, const float* __restrict__ WK,
    const float* __restrict__ WV, const float* __restrict__ Wg,
    const float* __restrict__ WO,
    bf16* __restrict__ WQb, bf16* __restrict__ WKgb,
    bf16* __restrict__ WVb, bf16* __restrict__ WOb) {
  int t = blockIdx.x * 256 + threadIdx.x;
  if (t >= DD * DD) return;
  int d = t >> 9;
  int e = t & 511;
  int h = e >> 6;
  int j = e & 63;
  const float* wk = WK + (size_t)d * DD + h * DH;
  float acc = 0.f;
#pragma unroll 8
  for (int i = 0; i < DH; ++i) acc += wk[i] * Wg[i * DH + j];
  WKgb[t] = (bf16)acc;
  WQb[t]  = (bf16)WQ[t];
  WVb[t]  = (bf16)WV[t];
  WOb[t]  = (bf16)WO[t];
}

// ------------------------------------------------------------------
// Generic WMMA GEMM: C[M=8192, N=512] = A[M,512] x W[512,512]
// Block: 256 threads = 8 waves. Block tile 128(M) x 64(N).
// MODE 0: row-major OT output [M,512]
// MODE 1: bf16 output pre-transposed per head: [B, H, DH, S]
// ------------------------------------------------------------------
template <typename AT, typename OT, int MODE>
__global__ __launch_bounds__(256) void gemm_tile(
    const AT* __restrict__ A, const bf16* __restrict__ Wb,
    OT* __restrict__ Cout) {
  __shared__ bf16 At[128][40];   // [m][k] padded
  __shared__ bf16 WtT[64][40];   // [n][k] padded (transposed W tile)

  const int tid  = threadIdx.x;
  const int wave = tid >> 5;
  const int lane = tid & 31;
  const int lm   = lane & 15;
  const int half = lane >> 4;
  const int h8   = half * 8;
  const int m0   = blockIdx.x * 128;
  const int n0   = blockIdx.y * 64;

  // staging coordinates (hoisted)
  const int ar = tid >> 1;            // A row 0..127
  const int ac = (tid & 1) * 16;      // A col 0 / 16
  const int wr = tid >> 3;            // W row 0..31
  const int wc = (tid & 7) * 8;       // W col 0..56

  v8f acc[4] = {};

  for (int k0 = 0; k0 < DD; k0 += 32) {
    __syncthreads();
    // stage A tile 128x32 -> bf16 LDS (through VGPRs: f32->bf16 convert)
    {
      const AT* src = A + (size_t)(m0 + ar) * DD + k0 + ac;
#pragma unroll
      for (int i = 0; i < 16; ++i) At[ar][ac + i] = (bf16)src[i];
      if (k0 + 32 < DD) __builtin_prefetch(src + 32, 0, 3);
    }
    // stage W tile 32x64 transposed -> WtT
    {
      const bf16* src = Wb + (size_t)(k0 + wr) * DD + n0 + wc;
#pragma unroll
      for (int i = 0; i < 8; ++i) WtT[wc + i][wr] = src[i];
      if (k0 + 32 < DD) __builtin_prefetch(src + 32 * DD, 0, 3);
    }
    __syncthreads();

    // A fragment: lane row = wave*16 + lm; k runs [h8,h8+8) and [16+h8,..)
    v16bf af;
    {
      const bf16* base = &At[wave * 16 + lm][0];
#pragma unroll
      for (int i = 0; i < 8; ++i) af[i] = base[h8 + i];
#pragma unroll
      for (int i = 0; i < 8; ++i) af[8 + i] = base[16 + h8 + i];
    }
#pragma unroll
    for (int nt = 0; nt < 4; ++nt) {
      v16bf bfr;
      const bf16* base = &WtT[nt * 16 + lm][0];
#pragma unroll
      for (int i = 0; i < 8; ++i) bfr[i] = base[h8 + i];
#pragma unroll
      for (int i = 0; i < 8; ++i) bfr[8 + i] = base[16 + h8 + i];
      acc[nt] = wmma_bf16(af, bfr, acc[nt]);
    }
  }

  // Write out. D layout: lane n = lm, row m = v + half*8
  if (MODE == 0) {
#pragma unroll
    for (int nt = 0; nt < 4; ++nt) {
#pragma unroll
      for (int v = 0; v < 8; ++v) {
        size_t m = (size_t)(m0 + wave * 16 + half * 8 + v);
        size_t n = (size_t)(n0 + nt * 16 + lm);
        Cout[m * DD + n] = (OT)acc[nt][v];
      }
    }
  } else {
    // transposed per-head layout [B, H, DH, S] for async V staging
    bf16* CT = (bf16*)Cout;
#pragma unroll
    for (int nt = 0; nt < 4; ++nt) {
      int n = n0 + nt * 16 + lm;         // global feature 0..511
      int hh = n >> 6;                   // head
      int dd = n & 63;                   // dim within head
      int mg = m0 + wave * 16 + half * 8;
      int bb = mg >> 11;                 // batch (S = 2048)
      int s  = mg & 2047;
      bf16* dst = CT + (((size_t)(bb * HH + hh) * DH + dd) * SS + s);
#pragma unroll
      for (int v = 0; v < 8; ++v) dst[v] = (bf16)acc[nt][v];
    }
  }
}

// ------------------------------------------------------------------
// Flash attention per (b, h). Block = 8 waves handles 128 q rows.
// Double-buffered async-to-LDS staging of K and (pre-transposed) V:
// tile t+1 copies are in flight while tile t is computed (ASYNCcnt
// pipelining). Softmax runs in A-fragment layout after one LDS bounce.
// ------------------------------------------------------------------
__global__ __launch_bounds__(256) void attn_kernel(
    const bf16* __restrict__ Ql, const bf16* __restrict__ Khw,
    const bf16* __restrict__ VlT, bf16* __restrict__ Hd) {
  const int blk  = blockIdx.x;          // B*H*(S/128) = 512
  const int qt   = blk & 15;            // S/128 = 16
  const int h    = (blk >> 4) & 7;
  const int b    = blk >> 7;
  const int q0   = qt * 128;
  const int hoff = h * DH;

  const int tid  = threadIdx.x;
  const int wave = tid >> 5;
  const int lane = tid & 31;
  const int lm   = lane & 15;
  const int half = lane >> 4;
  const int h8   = half * 8;

  __shared__ bf16  Kt[2][32][72];    // double-buffered [kv_row][d]
  __shared__ bf16  VtT[2][64][40];   // double-buffered [d][kv_row]
  __shared__ float St[8][16][36];    // per-wave raw scores [m][kv]

  const size_t rowbase = (size_t)b * SS * DD;
  const size_t vbase   = ((size_t)(b * HH + h) * DH) * SS;

  // async staging coordinates: 16 bytes per thread per tile
  const int kr = tid >> 3;           // K tile row 0..31
  const int kc = (tid & 7) * 8;      // K tile col
  const int vr = tid >> 2;           // V tile row (d) 0..63
  const int vc = (tid & 3) * 8;      // V tile col (kv)
  const unsigned ktDst[2] = {lds_addr_of(&Kt[0][kr][kc]),
                             lds_addr_of(&Kt[1][kr][kc])};
  const unsigned vtDst[2] = {lds_addr_of(&VtT[0][vr][vc]),
                             lds_addr_of(&VtT[1][vr][vc])};
  const bf16* kSrc = Khw + rowbase + (size_t)kr * DD + hoff + kc;
  const bf16* vSrc = VlT + vbase + (size_t)vr * SS + vc;

  // prologue: start tile 0 copies before anything else
  async_ld16(ktDst[0], kSrc);
  async_ld16(vtDst[0], vSrc);

  // Q fragments (16x64, two K=32 chunks), loaded once (overlaps async)
  v16bf aq[2];
  {
    const bf16* qrow =
        Ql + rowbase + (size_t)(q0 + wave * 16 + lm) * DD + hoff;
#pragma unroll
    for (int c = 0; c < 2; ++c) {
#pragma unroll
      for (int i = 0; i < 8; ++i) aq[c][i] = qrow[c * 32 + h8 + i];
#pragma unroll
      for (int i = 0; i < 8; ++i) aq[c][8 + i] = qrow[c * 32 + 16 + h8 + i];
    }
  }

  v8f acc[4] = {};
  float mrun = -1e30f, lrun = 0.f;   // per-lane: running max/sum of row lm

  const int NT = SS / 32;            // 64 KV tiles
  for (int t = 0; t < NT; ++t) {
    const int buf = t & 1;
    // all waves done reading buf^1 (tile t-1) before we overwrite it
    __syncthreads();
    if (t + 1 < NT) {
      // kick off tile t+1 into the other buffer, then wait only for
      // tile t's two copies (in-order completion => asynccnt <= 2)
      async_ld16(ktDst[buf ^ 1], kSrc + (size_t)(t + 1) * 32 * DD);
      async_ld16(vtDst[buf ^ 1], vSrc + (size_t)(t + 1) * 32);
      wait_asynccnt2();
    } else {
      wait_asynccnt0();
    }
    __syncthreads();

    // scores: s0 -> kv rows 0..15, s1 -> kv rows 16..31 (D layout)
    v8f s0 = {}, s1 = {};
#pragma unroll
    for (int c = 0; c < 2; ++c) {
      v16bf bk0, bk1;
      const bf16* r0 = &Kt[buf][lm][0];
      const bf16* r1 = &Kt[buf][16 + lm][0];
#pragma unroll
      for (int i = 0; i < 8; ++i) {
        bk0[i] = r0[c * 32 + h8 + i];
        bk1[i] = r1[c * 32 + h8 + i];
      }
#pragma unroll
      for (int i = 0; i < 8; ++i) {
        bk0[8 + i] = r0[c * 32 + 16 + h8 + i];
        bk1[8 + i] = r1[c * 32 + 16 + h8 + i];
      }
      s0 = wmma_bf16(aq[c], bk0, s0);
      s1 = wmma_bf16(aq[c], bk1, s1);
    }

    // bounce scores D-layout -> A-layout through per-wave LDS (f32)
#pragma unroll
    for (int v = 0; v < 8; ++v) {
      St[wave][half * 8 + v][lm]      = s0[v];
      St[wave][half * 8 + v][16 + lm] = s1[v];
    }
    float vals[16];
    {
      const float* row = &St[wave][lm][0];
#pragma unroll
      for (int i = 0; i < 8; ++i) vals[i] = row[h8 + i];
#pragma unroll
      for (int i = 0; i < 8; ++i) vals[8 + i] = row[16 + h8 + i];
    }

    // row-local softmax: this lane owns half of row lm (partner = lane^16)
    float mx = vals[0];
#pragma unroll
    for (int i = 1; i < 16; ++i) mx = fmaxf(mx, vals[i]);
    mx = fmaxf(mx, __shfl_xor(mx, 16, 32));      // combine halves
    float mnew = fmaxf(mrun, mx);
    float corr = __expf(mrun - mnew);
    mrun = mnew;

    v16bf ap;                                    // P in A-fragment layout
    float rs = 0.f;
#pragma unroll
    for (int i = 0; i < 16; ++i) {
      float e = __expf(vals[i] - mnew);
      rs += e;
      ap[i] = (bf16)e;
    }
    rs += __shfl_xor(rs, 16, 32);                // full row sum
    lrun = lrun * corr + rs;

    // broadcast per-row corr to D-layout accumulator rows (m = half*8+v)
#pragma unroll
    for (int v = 0; v < 8; ++v) {
      float cv = __shfl(corr, half * 8 + v, 32);
#pragma unroll
      for (int dt = 0; dt < 4; ++dt) acc[dt][v] *= cv;
    }

    // acc[dt] += P(16x32) x V(32x16)
#pragma unroll
    for (int dt = 0; dt < 4; ++dt) {
      v16bf bv;
      const bf16* vrow = &VtT[buf][dt * 16 + lm][0];
#pragma unroll
      for (int i = 0; i < 8; ++i) bv[i] = vrow[h8 + i];
#pragma unroll
      for (int i = 0; i < 8; ++i) bv[8 + i] = vrow[16 + h8 + i];
      acc[dt] = wmma_bf16(ap, bv, acc[dt]);
    }
  }

  // broadcast row sums to D-layout, normalize, store head output (bf16)
  float lD[8];
#pragma unroll
  for (int v = 0; v < 8; ++v) lD[v] = __shfl(lrun, half * 8 + v, 32);
#pragma unroll
  for (int dt = 0; dt < 4; ++dt) {
#pragma unroll
    for (int v = 0; v < 8; ++v) {
      float o = acc[dt][v] / lD[v];
      size_t m = (size_t)(q0 + wave * 16 + half * 8 + v);
      Hd[rowbase + m * DD + hoff + dt * 16 + lm] = (bf16)o;
    }
  }
}

// ------------------------------------------------------------------
extern "C" void kernel_launch(void* const* d_in, const int* in_sizes, int n_in,
                              void* d_out, int out_size, void* d_ws,
                              size_t ws_size, hipStream_t stream) {
  const float* Q  = (const float*)d_in[0];
  const float* K  = (const float*)d_in[1];
  const float* V  = (const float*)d_in[2];
  // d_in[3] == M, unused by the reference layer
  const float* WQ = (const float*)d_in[4];
  const float* WK = (const float*)d_in[5];
  const float* WV = (const float*)d_in[6];
  const float* Wg = (const float*)d_in[7];
  const float* WO = (const float*)d_in[8];
  float* out = (float*)d_out;

  // workspace carve-out (bf16 buffers)
  char* ws = (char*)d_ws;
  size_t off = 0;
  auto carve = [&](size_t bytes) {
    char* p = ws + off;
    off += (bytes + 255) & ~(size_t)255;
    return p;
  };
  const size_t wbytes = (size_t)DD * DD * sizeof(bf16);
  const size_t abytes = (size_t)MTOT * DD * sizeof(bf16);
  bf16* WQb  = (bf16*)carve(wbytes);
  bf16* WKgb = (bf16*)carve(wbytes);
  bf16* WVb  = (bf16*)carve(wbytes);
  bf16* WOb  = (bf16*)carve(wbytes);
  bf16* Ql   = (bf16*)carve(abytes);
  bf16* Khw  = (bf16*)carve(abytes);
  bf16* VlT  = (bf16*)carve(abytes);   // [B, H, DH, S]
  bf16* Hd   = (bf16*)carve(abytes);

  // 0) fold W_gen into W_K, convert weights to bf16
  prep_weights<<<dim3((DD * DD + 255) / 256), dim3(256), 0, stream>>>(
      WQ, WK, WV, Wg, WO, WQb, WKgb, WVb, WOb);

  // 1) projections (fp32 A -> bf16); V written pre-transposed per head
  dim3 ggrid(MTOT / 128, DD / 64);
  gemm_tile<float, bf16, 0><<<ggrid, dim3(256), 0, stream>>>(Q, WQb, Ql);
  gemm_tile<float, bf16, 0><<<ggrid, dim3(256), 0, stream>>>(K, WKgb, Khw);
  gemm_tile<float, bf16, 1><<<ggrid, dim3(256), 0, stream>>>(V, WVb, VlT);

  // 2) flash attention per (b, h, 128-q-row block)
  attn_kernel<<<dim3(BB * HH * (SS / 128)), dim3(256), 0, stream>>>(
      Ql, Khw, VlT, Hd);

  // 3) output projection (bf16 A -> fp32 out)
  gemm_tile<bf16, float, 0><<<ggrid, dim3(256), 0, stream>>>(Hd, WOb, out);
}
